// CGAtNet_21706764714724
// MI455X (gfx1250) — compile-verified, matrix operand order checked by reading, not run on hardware
//
#include <hip/hip_runtime.h>
#include <hip/hip_bf16.h>

typedef __bf16 bf16_t;
typedef __attribute__((ext_vector_type(4)))  __bf16 v4bf;
typedef __attribute__((ext_vector_type(16))) __bf16 v16bf;
typedef __attribute__((ext_vector_type(8)))  float  v8f;

#define NN 10000     // nodes
#define NE 320000    // edges
#define CD 64        // channels
#define NH 4         // heads
#define HD 128       // hidden per head

// ---------- helpers ----------
static __device__ __forceinline__ unsigned ford(float f) {
  unsigned u = __float_as_uint(f);
  return (u & 0x80000000u) ? ~u : (u | 0x80000000u);   // order-preserving encode
}
static __device__ __forceinline__ float fdec(unsigned e) {
  unsigned u = (e & 0x80000000u) ? (e ^ 0x80000000u) : ~e;
  return __uint_as_float(u);
}
static __device__ __forceinline__ float lrelu(float v) { return v > 0.f ? v : 0.01f * v; }
static __device__ __forceinline__ v4bf cvt4(float4 v) {
  v4bf o; o[0] = (bf16_t)v.x; o[1] = (bf16_t)v.y; o[2] = (bf16_t)v.z; o[3] = (bf16_t)v.w;
  return o;
}

// A-fragment (16x32 MxK, bf16) from an LDS row pointer. lane l: row = l%16,
// K(e) = kt*32 + (e/8)*16 + (l/16)*8 + (e%8)  ->  two contiguous 16B chunks.
static __device__ __forceinline__ v16bf load_a_lds(const bf16_t* row, int kt, int half) {
  union { v16bf v; uint4 q[2]; } u;
  const bf16_t* p = row + kt * 32 + half * 8;
  u.q[0] = *(const uint4*)(p);
  u.q[1] = *(const uint4*)(p + 16);
  return u.v;
}
// B-fragment from pre-packed weights: tile is 512 bf16, lane l reads 16 contiguous.
static __device__ __forceinline__ v16bf load_b_pk(const bf16_t* tile, int lane) {
  return ((const v16bf*)tile)[lane];
}
static __device__ __forceinline__ v8f wmma_bf16(v16bf a, v16bf b, v8f c) {
  return __builtin_amdgcn_wmma_f32_16x16x32_bf16(false, a, false, b, (short)0, c, false, false);
}

// ---------- weight packing: B[k][n] = W[n][k] into per-lane WMMA layout ----------
__global__ void pack_nk_kernel(const float* __restrict__ W, bf16_t* __restrict__ dst,
                               int NT, int KT, int ldk) {
  int t = blockIdx.x * 256 + threadIdx.x;
  if (t >= NT * KT * 512) return;
  int e = t & 15, l = (t >> 4) & 31, rest = t >> 9;
  int kt = rest % KT, nt = rest / KT;
  int n = nt * 16 + (l & 15);
  int k = kt * 32 + ((e >> 3) << 4) + ((l >> 4) << 3) + (e & 7);
  dst[t] = (bf16_t)W[(size_t)n * ldk + k];
}

__global__ void pack_w1_kernel(const float* __restrict__ Wa1, const float* __restrict__ Wm1,
                               bf16_t* __restrict__ dst) {
  int t = blockIdx.x * 256 + threadIdx.x;   // 64 ntiles * 6 ksteps * 512
  if (t >= 64 * 6 * 512) return;
  int e = t & 15, l = (t >> 4) & 31, rest = t >> 9;
  int kt = rest % 6, nt = rest / 6;
  int n = nt * 16 + (l & 15);                               // 0..1023: [Wa1 512 | Wm1 512]
  int k = kt * 32 + ((e >> 3) << 4) + ((l >> 4) << 3) + (e & 7);  // 0..191
  float v = (n < 512) ? Wa1[(size_t)n * 192 + k] : Wm1[(size_t)(n - 512) * 192 + k];
  dst[t] = (bf16_t)v;
}

__global__ void pack_wm2_kernel(const float* __restrict__ Wm2, bf16_t* __restrict__ dst) {
  int t = blockIdx.x * 256 + threadIdx.x;   // 4 heads * 4 ntiles * 4 ksteps * 512
  if (t >= 4 * 4 * 4 * 512) return;
  int e = t & 15, l = (t >> 4) & 31, rest = t >> 9;
  int kt = rest & 3, nt = (rest >> 2) & 3, h = rest >> 4;
  int n = nt * 16 + (l & 15);                               // 0..63
  int k = kt * 32 + ((e >> 3) << 4) + ((l >> 4) << 3) + (e & 7);  // 0..127
  dst[t] = (bf16_t)Wm2[((size_t)h * 64 + n) * 128 + k];
}

__global__ void pack_bias1_kernel(const float* __restrict__ ba1, const float* __restrict__ bm1,
                                  float* __restrict__ dst) {
  int t = blockIdx.x * 256 + threadIdx.x;
  if (t >= 1024) return;
  dst[t] = (t < 512) ? ba1[t] : bm1[t - 512];
}

// ---------- edge MLP: e = lrelu(ea @ We1^T + be1) @ We2^T + be2 ----------
__global__ void __launch_bounds__(256) edge_mlp_kernel(
    const float* __restrict__ ea,
    const bf16_t* __restrict__ pWe1, const float* __restrict__ be1,
    const bf16_t* __restrict__ pWe2, const float* __restrict__ be2,
    bf16_t* __restrict__ e_ws) {
  __shared__ __align__(16) bf16_t A1[128][72];
  __shared__ __align__(16) bf16_t H1[128][72];
  const int tid = threadIdx.x;
  const size_t e0 = (size_t)blockIdx.x * 128;
  for (int i = tid; i < 128 * 16; i += 256) {   // float4 quads: 128 rows x 16 quads
    int r = i >> 4, c4 = (i & 15) << 2;
    float4 v = *(const float4*)&ea[(e0 + r) * CD + c4];
    *(v4bf*)&A1[r][c4] = cvt4(v);
  }
  __syncthreads();
  const int wave = tid >> 5, lane = tid & 31;
  const int half = lane >> 4, nl = lane & 15;
  const int r0 = wave << 4;
  for (int nt = 0; nt < 4; ++nt) {
    v8f acc = {};
    for (int kt = 0; kt < 2; ++kt) {
      v16bf a = load_a_lds(&A1[r0 + nl][0], kt, half);
      v16bf b = load_b_pk(pWe1 + (size_t)(nt * 2 + kt) * 512, lane);
      acc = wmma_bf16(a, b, acc);
    }
    float bc = be1[nt * 16 + nl];
    for (int vi = 0; vi < 8; ++vi)
      H1[r0 + vi + half * 8][nt * 16 + nl] = (bf16_t)lrelu(acc[vi] + bc);
  }
  __syncthreads();
  for (int nt = 0; nt < 4; ++nt) {
    v8f acc = {};
    for (int kt = 0; kt < 2; ++kt) {
      v16bf a = load_a_lds(&H1[r0 + nl][0], kt, half);
      v16bf b = load_b_pk(pWe2 + (size_t)(nt * 2 + kt) * 512, lane);
      acc = wmma_bf16(a, b, acc);
    }
    float bc = be2[nt * 16 + nl];
    for (int vi = 0; vi < 8; ++vi)
      e_ws[(e0 + r0 + vi + half * 8) * CD + nt * 16 + nl] = (bf16_t)(acc[vi] + bc);
  }
}

// ---------- main fused kernel: 64 edges/block, [64x192]@[192x1024] + heads ----------
// Each B fragment is loaded once and reused across 4 M-tiles (4 independent
// accumulators -> back-to-back WMMAs, no D->A/B hazards, 4x less L2 weight traffic).
__global__ void __launch_bounds__(256) gat_main_kernel(
    const float* __restrict__ x, const int* __restrict__ ei,
    const bf16_t* __restrict__ e_ws,
    const bf16_t* __restrict__ pW1, const float* __restrict__ bias1,
    const float* __restrict__ Wa2, const float* __restrict__ ba2,
    const bf16_t* __restrict__ pWm2, const float* __restrict__ bm2,
    float* __restrict__ alpha_ws, bf16_t* __restrict__ msg_ws,
    unsigned* __restrict__ nodemax) {
  __shared__ __align__(16) bf16_t mt[64][200];    // m = [x_i | e | x_j]
  __shared__ __align__(16) bf16_t hid[64][1040];  // [A-net 512 | M-net 512]
  __shared__ int sdst[64];
  const int tid = threadIdx.x;
  const int e0 = blockIdx.x * 64;
  if (tid < 64) sdst[tid] = ei[NE + e0 + tid];
  // stage m = [x_i | e | x_j] as 4-channel quads: 64 rows x 48 quads, 12 iters
  for (int i = tid; i < 64 * 48; i += 256) {
    int r = i / 48, q = i - r * 48;
    int sec = q >> 4, c4 = (q & 15) << 2;
    int eg = e0 + r;
    v4bf o;
    if (sec == 1) {
      o = *(const v4bf*)&e_ws[(size_t)eg * CD + c4];
    } else {
      int node = (sec == 0) ? ei[NE + eg] : ei[eg];   // x_i = x[dst], x_j = x[src]
      o = cvt4(*(const float4*)&x[(size_t)node * CD + c4]);
    }
    *(v4bf*)&mt[r][sec * 64 + c4] = o;
  }
  __syncthreads();
  const int wave = tid >> 5, lane = tid & 31;
  const int half = lane >> 4, nl = lane & 15;
  // layer 1: each wave covers 8 of the 64 N-tiles x 4 M-tiles, K = 192 (6 k-steps)
  for (int i = 0; i < 8; ++i) {
    int nt = wave + (i << 3);
    v8f acc0 = {}, acc1 = {}, acc2 = {}, acc3 = {};
    for (int kt = 0; kt < 6; ++kt) {
      v16bf b = load_b_pk(pW1 + (size_t)(nt * 6 + kt) * 512, lane);
      v16bf a0 = load_a_lds(&mt[nl][0], kt, half);
      acc0 = wmma_bf16(a0, b, acc0);
      v16bf a1 = load_a_lds(&mt[16 + nl][0], kt, half);
      acc1 = wmma_bf16(a1, b, acc1);
      v16bf a2 = load_a_lds(&mt[32 + nl][0], kt, half);
      acc2 = wmma_bf16(a2, b, acc2);
      v16bf a3 = load_a_lds(&mt[48 + nl][0], kt, half);
      acc3 = wmma_bf16(a3, b, acc3);
    }
    float bc = bias1[nt * 16 + nl];
    for (int vi = 0; vi < 8; ++vi) {
      int m = vi + half * 8;
      hid[m][nt * 16 + nl]      = (bf16_t)lrelu(acc0[vi] + bc);
      hid[16 + m][nt * 16 + nl] = (bf16_t)lrelu(acc1[vi] + bc);
      hid[32 + m][nt * 16 + nl] = (bf16_t)lrelu(acc2[vi] + bc);
      hid[48 + m][nt * 16 + nl] = (bf16_t)lrelu(acc3[vi] + bc);
    }
  }
  __syncthreads();
  // layer 2 (message): per head [64x128] @ [128x64], 16 (h,nt) jobs over 8 waves
  for (int i = 0; i < 2; ++i) {
    int job = (wave << 1) | i;
    int h = job >> 2, nt = job & 3;
    v8f acc0 = {}, acc1 = {}, acc2 = {}, acc3 = {};
    for (int kt = 0; kt < 4; ++kt) {
      v16bf b = load_b_pk(pWm2 + (size_t)((h * 4 + nt) * 4 + kt) * 512, lane);
      v16bf a0 = load_a_lds(&hid[nl][512 + h * HD], kt, half);
      acc0 = wmma_bf16(a0, b, acc0);
      v16bf a1 = load_a_lds(&hid[16 + nl][512 + h * HD], kt, half);
      acc1 = wmma_bf16(a1, b, acc1);
      v16bf a2 = load_a_lds(&hid[32 + nl][512 + h * HD], kt, half);
      acc2 = wmma_bf16(a2, b, acc2);
      v16bf a3 = load_a_lds(&hid[48 + nl][512 + h * HD], kt, half);
      acc3 = wmma_bf16(a3, b, acc3);
    }
    float bc = bm2[h * CD + nt * 16 + nl];
    for (int vi = 0; vi < 8; ++vi) {
      int m = vi + half * 8;
      msg_ws[((size_t)(e0 + m) * NH + h) * CD + nt * 16 + nl]      = (bf16_t)(acc0[vi] + bc);
      msg_ws[((size_t)(e0 + 16 + m) * NH + h) * CD + nt * 16 + nl] = (bf16_t)(acc1[vi] + bc);
      msg_ws[((size_t)(e0 + 32 + m) * NH + h) * CD + nt * 16 + nl] = (bf16_t)(acc2[vi] + bc);
      msg_ws[((size_t)(e0 + 48 + m) * NH + h) * CD + nt * 16 + nl] = (bf16_t)(acc3[vi] + bc);
    }
  }
  // layer 2 (attention logit): vectorized 128-dot per (edge,head) + stable-max scatter
  {
    int r = tid >> 2, h = tid & 3;   // 64 rows x 4 heads == 256 threads
    float s = ba2[h];
    const v4bf* hp = (const v4bf*)&hid[r][h * HD];
    const float4* wp = (const float4*)&Wa2[h * HD];
#pragma unroll 4
    for (int k = 0; k < HD / 4; ++k) {
      v4bf hv = hp[k];
      float4 wv = wp[k];
      s += (float)hv[0] * wv.x + (float)hv[1] * wv.y + (float)hv[2] * wv.z + (float)hv[3] * wv.w;
    }
    alpha_ws[(size_t)(e0 + r) * NH + h] = s;
    atomicMax(&nodemax[(size_t)sdst[r] * NH + h], ford(s));
  }
}

// ---------- softmax numerator/denominator scatter ----------
__global__ void __launch_bounds__(256) scatter_kernel(
    const int* __restrict__ ei, const float* __restrict__ alpha_ws,
    const bf16_t* __restrict__ msg_ws, const unsigned* __restrict__ nodemax,
    float* __restrict__ den, float* __restrict__ num) {
  int t = blockIdx.x * 256 + threadIdx.x;   // NE * NH * 4 threads, 16 channels each
  if (t >= NE * 16) return;
  int cg = t & 3, h = (t >> 2) & 3, e = t >> 4;
  int d = ei[NE + e];
  float a = alpha_ws[(size_t)e * NH + h];
  float ex = __expf(a - fdec(nodemax[(size_t)d * NH + h]));
  if (cg == 0) atomicAdd(&den[(size_t)d * NH + h], ex);
  const bf16_t* mp = msg_ws + ((size_t)e * NH + h) * CD + cg * 16;
  float* np = num + ((size_t)d * NH + h) * CD + cg * 16;
#pragma unroll
  for (int c = 0; c < 16; ++c) atomicAdd(np + c, ex * (float)mp[c]);
}

// ---------- out = mean over heads of num/den ----------
__global__ void __launch_bounds__(256) finalize_kernel(
    const float* __restrict__ num, const float* __restrict__ den, float* __restrict__ out) {
  int t = blockIdx.x * 256 + threadIdx.x;
  if (t >= NN * CD) return;
  int n = t >> 6, c = t & 63;
  float s = 0.f;
#pragma unroll
  for (int h = 0; h < NH; ++h) {
    float d = den[n * NH + h];
    float v = num[(size_t)(n * NH + h) * CD + c];
    s += (d > 0.f) ? v / d : 0.f;
  }
  out[t] = s * 0.25f;
}

extern "C" void kernel_launch(void* const* d_in, const int* in_sizes, int n_in,
                              void* d_out, int out_size, void* d_ws, size_t ws_size,
                              hipStream_t stream) {
  (void)in_sizes; (void)n_in; (void)out_size; (void)ws_size;
  const float* x   = (const float*)d_in[0];
  const int*   ei  = (const int*)d_in[1];
  const float* ea  = (const float*)d_in[2];
  const float* We1 = (const float*)d_in[3];
  const float* be1 = (const float*)d_in[4];
  const float* We2 = (const float*)d_in[5];
  const float* be2 = (const float*)d_in[6];
  const float* Wa1 = (const float*)d_in[7];
  const float* ba1 = (const float*)d_in[8];
  const float* Wa2 = (const float*)d_in[9];
  const float* ba2 = (const float*)d_in[10];
  const float* Wm1 = (const float*)d_in[11];
  const float* bm1 = (const float*)d_in[12];
  const float* Wm2 = (const float*)d_in[13];
  const float* bm2 = (const float*)d_in[14];
  float* out = (float*)d_out;

  char* ws = (char*)d_ws;
  size_t off = 0;
  auto alloc = [&](size_t bytes) -> void* {
    void* p = ws + off;
    off += (bytes + 255) & ~(size_t)255;
    return p;
  };
  bf16_t* pWe1  = (bf16_t*)alloc((size_t)4 * 2 * 512 * 2);
  bf16_t* pWe2  = (bf16_t*)alloc((size_t)4 * 2 * 512 * 2);
  bf16_t* pW1   = (bf16_t*)alloc((size_t)64 * 6 * 512 * 2);
  bf16_t* pWm2  = (bf16_t*)alloc((size_t)64 * 512 * 2);
  float*  bias1 = (float*)alloc(1024 * 4);
  bf16_t* e_ws  = (bf16_t*)alloc((size_t)NE * CD * 2);
  float*  alpha_ws = (float*)alloc((size_t)NE * NH * 4);
  bf16_t* msg_ws   = (bf16_t*)alloc((size_t)NE * NH * CD * 2);
  unsigned* nodemax = (unsigned*)alloc((size_t)NN * NH * 4);
  float* den = (float*)alloc((size_t)NN * NH * 4);
  float* num = (float*)alloc((size_t)NN * NH * CD * 4);

  hipMemsetAsync(nodemax, 0, (size_t)NN * NH * 4, stream);  // ordered-encode min
  hipMemsetAsync(den, 0, (size_t)NN * NH * 4, stream);
  hipMemsetAsync(num, 0, (size_t)NN * NH * CD * 4, stream);

  pack_nk_kernel<<<16, 256, 0, stream>>>(We1, pWe1, 4, 2, 64);
  pack_nk_kernel<<<16, 256, 0, stream>>>(We2, pWe2, 4, 2, 64);
  pack_w1_kernel<<<768, 256, 0, stream>>>(Wa1, Wm1, pW1);
  pack_wm2_kernel<<<128, 256, 0, stream>>>(Wm2, pWm2);
  pack_bias1_kernel<<<4, 256, 0, stream>>>(ba1, bm1, bias1);

  edge_mlp_kernel<<<NE / 128, 256, 0, stream>>>(ea, pWe1, be1, pWe2, be2, e_ws);
  gat_main_kernel<<<NE / 64, 256, 0, stream>>>(x, ei, e_ws, pW1, bias1, Wa2, ba2,
                                               pWm2, bm2, alpha_ws, msg_ws, nodemax);
  scatter_kernel<<<(NE * 16) / 256, 256, 0, stream>>>(ei, alpha_ws, msg_ws, nodemax, den, num);
  finalize_kernel<<<(NN * CD + 255) / 256, 256, 0, stream>>>(num, den, out);
}
